// NSM_50852412785381
// MI455X (gfx1250) — matrix operand order repaired and checked by reading, not compiled, not run on hardware
//
#include <hip/hip_runtime.h>

// ---------------------------------------------------------------------------
// y = x @ W^T, then 100x: y = xW^T - y@M   ==>   y = x @ (W^T * S),
// S = sum_{j=0}^{100} (-M)^j  (== sum to 128 terms in f32, rho(M)~0.5).
// Stage 1: S via 7 log-doubling steps (tiny).  Stage 2: C = W^T S (tiny).
// Stage 3: one big memory-bound GEMM y = x@C using bf16 hi/lo split WMMA.
// ---------------------------------------------------------------------------

typedef __attribute__((ext_vector_type(16))) __bf16 v16bf;
typedef __attribute__((ext_vector_type(8)))  __bf16 v8bf;
typedef __attribute__((ext_vector_type(8)))  float  v8f;
typedef __attribute__((ext_vector_type(4)))  float  v4f;

constexpr int IN   = 512;
constexpr int OUT  = 256;
constexpr int BM   = 128;   // rows per block
constexpr int BK   = 32;    // K chunk (bf16 WMMA K)
constexpr int PAD  = 40;    // LDS inner stride (80B: 16B-aligned frag loads)

// ---- Stage 1a: S = I, P = -M --------------------------------------------
__global__ __launch_bounds__(256) void nsm_init_SP(const float* __restrict__ M,
                                                   float* __restrict__ S,
                                                   float* __restrict__ P) {
  int i = blockIdx.x * 256 + threadIdx.x;       // 65536 elements
  int r = i >> 8, c = i & 255;
  P[i] = -M[i];
  S[i] = (r == c) ? 1.0f : 0.0f;
}

// ---- Stage 1b: one doubling step: S' = S + P@S ; P' = P@P ---------------
__global__ __launch_bounds__(256) void nsm_double(const float* __restrict__ Sin,
                                                  const float* __restrict__ Pin,
                                                  float* __restrict__ Sout,
                                                  float* __restrict__ Pout) {
  const int m = blockIdx.x;    // 256
  const int n = threadIdx.x;   // 256
  float accS = Sin[m * 256 + n];
  float accP = 0.0f;
  for (int k = 0; k < 256; ++k) {
    float p = Pin[m * 256 + k];
    accS = fmaf(p, Sin[k * 256 + n], accS);
    accP = fmaf(p, Pin[k * 256 + n], accP);
  }
  Sout[m * 256 + n] = accS;
  Pout[m * 256 + n] = accP;
}

// ---- Stage 2: C[k][n] = sum_m W[m][k] * S[m][n]   (C = W^T S, 512x256) ---
__global__ __launch_bounds__(256) void nsm_make_C(const float* __restrict__ W,
                                                  const float* __restrict__ S,
                                                  float* __restrict__ C) {
  const int k = blockIdx.x;    // 512
  const int n = threadIdx.x;   // 256
  float acc = 0.0f;
  for (int m = 0; m < 256; ++m)
    acc = fmaf(W[m * 512 + k], S[m * 256 + n], acc);
  C[k * 256 + n] = acc;
}

// ---- Stage 3: y[B,256] = x[B,512] @ C[512,256], bf16 hi/lo split WMMA ----
__global__ __launch_bounds__(256) void nsm_gemm(const float* __restrict__ x,
                                                const float* __restrict__ C,
                                                float* __restrict__ y) {
  __shared__ __bf16 Ah[BM][PAD];
  __shared__ __bf16 Al[BM][PAD];
  __shared__ __bf16 Bh[OUT][PAD];   // stored transposed: [n][k]
  __shared__ __bf16 Bl[OUT][PAD];

  const int t    = threadIdx.x;
  const int wave = t >> 5;          // 0..7
  const int lane = t & 31;
  const int sub  = lane & 15;
  const int hi   = lane >> 4;       // 0 or 1 (lane half)
  const long rowBase = (long)blockIdx.x * BM;

  v8f acc[16];
#pragma unroll
  for (int i = 0; i < 16; ++i) {
    v8f z = {0.f, 0.f, 0.f, 0.f, 0.f, 0.f, 0.f, 0.f};
    acc[i] = z;
  }

  for (int kc = 0; kc < IN / BK; ++kc) {
    const int k0 = kc * BK;

    // stage x tile: 128 rows x 32 k (f32 -> bf16 hi/lo), [row][k]
    {
      const int row  = t >> 1;
      const int half = (t & 1) * 16;
      const float* src = x + (rowBase + row) * IN + k0 + half;
#pragma unroll
      for (int j = 0; j < 16; j += 4) {
        v4f v = *(const v4f*)(src + j);            // global_load_b128
#pragma unroll
        for (int e = 0; e < 4; ++e) {
          float f = v[e];
          __bf16 h = (__bf16)f;
          __bf16 l = (__bf16)(f - (float)h);
          Ah[row][half + j + e] = h;
          Al[row][half + j + e] = l;
        }
      }
    }
    // stage C chunk: 32 k x 256 n, transposed to [n][k] (f32 -> hi/lo)
    {
      const int n = t;
#pragma unroll 4
      for (int kk = 0; kk < BK; ++kk) {
        float f = C[(k0 + kk) * OUT + n];          // coalesced over t
        __bf16 h = (__bf16)f;
        __bf16 l = (__bf16)(f - (float)h);
        Bh[n][kk] = h;
        Bl[n][kk] = l;
      }
    }
    __syncthreads();

    // A fragments (ISA 16-bit A 16x32 layout):
    //   lanes 0-15 : K[0..7] ++ K[16..23];  lanes 16-31: K[8..15] ++ K[24..31]
    const int arow = wave * 16 + sub;
    const int g0 = hi ? 8 : 0;
    const int g1 = hi ? 24 : 16;
    v8bf ah0 = *(const v8bf*)(&Ah[arow][g0]);
    v8bf ah1 = *(const v8bf*)(&Ah[arow][g1]);
    v8bf al0 = *(const v8bf*)(&Al[arow][g0]);
    v8bf al1 = *(const v8bf*)(&Al[arow][g1]);
    v16bf a_h = __builtin_shufflevector(ah0, ah1, 0,1,2,3,4,5,6,7,8,9,10,11,12,13,14,15);
    v16bf a_l = __builtin_shufflevector(al0, al1, 0,1,2,3,4,5,6,7,8,9,10,11,12,13,14,15);

    // B fragments (dense bf16 B 32x16): lanes 0-15 hold K[0..7] ++ K[16..23] of col n,
    // lanes 16-31 hold K[8..15] ++ K[24..31] (same order as A) in transposed [n][k] storage.
#pragma unroll
    for (int tile = 0; tile < 16; ++tile) {
      const int col = tile * 16 + sub;
      const __bf16* bph = &Bh[col][hi * 8];
      const __bf16* bpl = &Bl[col][hi * 8];
      v8bf bh0 = *(const v8bf*)(bph);
      v8bf bh1 = *(const v8bf*)(bph + 16);
      v8bf bl0 = *(const v8bf*)(bpl);
      v8bf bl1 = *(const v8bf*)(bpl + 16);
      v16bf b_h = __builtin_shufflevector(bh0, bh1, 0,1,2,3,4,5,6,7,8,9,10,11,12,13,14,15);
      v16bf b_l = __builtin_shufflevector(bl0, bl1, 0,1,2,3,4,5,6,7,8,9,10,11,12,13,14,15);

      // acc += xh*Ch + xh*Cl + xl*Ch   (f32 accumulate)
      acc[tile] = __builtin_amdgcn_wmma_f32_16x16x32_bf16(
          false, a_h, false, b_h, (short)0, acc[tile], false, false);
      acc[tile] = __builtin_amdgcn_wmma_f32_16x16x32_bf16(
          false, a_h, false, b_l, (short)0, acc[tile], false, false);
      acc[tile] = __builtin_amdgcn_wmma_f32_16x16x32_bf16(
          false, a_l, false, b_h, (short)0, acc[tile], false, false);
    }
    __syncthreads();
  }

  // epilogue: C/D layout — VGPR i: lanes0-15 -> M=i, lanes16-31 -> M=8+i
#pragma unroll
  for (int tile = 0; tile < 16; ++tile) {
    const int n = tile * 16 + sub;
    const long r0 = rowBase + wave * 16 + hi * 8;
#pragma unroll
    for (int i = 0; i < 8; ++i)
      y[(r0 + i) * OUT + n] = acc[tile][i];
  }
}

// ---------------------------------------------------------------------------
extern "C" void kernel_launch(void* const* d_in, const int* in_sizes, int n_in,
                              void* d_out, int out_size, void* d_ws, size_t ws_size,
                              hipStream_t stream) {
  const float* x = (const float*)d_in[0];   // [65536, 512]
  const float* W = (const float*)d_in[1];   // [256, 512]
  const float* M = (const float*)d_in[2];   // [256, 256]
  float* y = (float*)d_out;                 // [65536, 256]

  float* ws = (float*)d_ws;
  float* Sbuf[2] = {ws,                ws + 2 * 65536};
  float* Pbuf[2] = {ws + 1 * 65536,    ws + 3 * 65536};
  float* C       = ws + 4 * 65536;          // 512*256 floats

  // S = I, P = -M
  nsm_init_SP<<<256, 256, 0, stream>>>(M, Sbuf[0], Pbuf[0]);
  // 7 doublings: S covers 2^7 = 128 terms (>= 101; tail < f32 eps)
  for (int s = 0; s < 7; ++s) {
    nsm_double<<<256, 256, 0, stream>>>(Sbuf[s & 1], Pbuf[s & 1],
                                        Sbuf[(s + 1) & 1], Pbuf[(s + 1) & 1]);
  }
  // C = W^T S  (final S is in buffer 7&1 == 1)
  nsm_make_C<<<512, 256, 0, stream>>>(W, Sbuf[1], C);
  // y = x @ C   (memory-bound: ~192 MB -> ~8 us at 23.3 TB/s)
  nsm_gemm<<<65536 / BM, 256, 0, stream>>>(x, C, y);
}
